// LSTM_65833258713392
// MI455X (gfx1250) — matrix operand order repaired and checked by reading, not compiled, not run on hardware
//
#include <hip/hip_runtime.h>
#include <hip/hip_bf16.h>

typedef __attribute__((ext_vector_type(16))) _Float16 v16h;
typedef __attribute__((ext_vector_type(8)))  _Float16 v8h;
typedef __attribute__((ext_vector_type(4)))  _Float16 v4h;
typedef __attribute__((ext_vector_type(8)))  float    v8f;
typedef __attribute__((ext_vector_type(4)))  float    v4f;

#define SEQ   512
#define BATCH 64
#define HID   1024
#define GATES 4096          // 4*HID
#define MTOT  32768         // SEQ*BATCH
#define LDSPITCH 1032       // 1024 + 8 halves pad -> 4-bank rotation per column

// ---------------------------------------------------------------------------
// WMMA helpers (CDNA5 16x16x32 f16 -> f32)
// ---------------------------------------------------------------------------
__device__ __forceinline__ v8f wmma_f16(v16h a, v16h b, v8f c) {
    return __builtin_amdgcn_wmma_f32_16x16x32_f16(false, a, false, b,
                                                  (short)0, c, false, false);
}

// A-fragment: lane l holds row (l&15); elements 0..15 = K {k0..k0+15},
// k0 = kc + 16*(l>>4). Two 16B loads.
__device__ __forceinline__ v16h load_a_frag(const _Float16* p) {
    v8h lo = *(const v8h*)(p);
    v8h hi = *(const v8h*)(p + 8);
    return __builtin_shufflevector(lo, hi, 0,1,2,3,4,5,6,7,8,9,10,11,12,13,14,15);
}

// B-fragment from LDS (16B-aligned only, so two v8h ds loads).
__device__ __forceinline__ v16h load_b_frag_lds(const _Float16* p) {
    v8h lo = *(const v8h*)(p);
    v8h hi = *(const v8h*)(p + 8);
    return __builtin_shufflevector(lo, hi, 0,1,2,3,4,5,6,7,8,9,10,11,12,13,14,15);
}

__device__ __forceinline__ float sigmoid_f(float x) {
    return 1.0f / (1.0f + __expf(-x));
}
__device__ __forceinline__ float tanh_f(float x) {
    float ax = fabsf(x);
    float e  = __expf(2.0f * ax);            // overflow -> +inf -> r = 1
    float r  = 1.0f - 2.0f / (e + 1.0f);
    return copysignf(r, x);
}

// ---------------------------------------------------------------------------
// f32 -> f16 conversion (n multiple of 4)
// ---------------------------------------------------------------------------
__global__ __launch_bounds__(256) void cvt_f32_to_f16(const float* __restrict__ in,
                                                      _Float16* __restrict__ out,
                                                      int n) {
    int i = (blockIdx.x * 256 + threadIdx.x) * 4;
    if (i < n) {
        v4f v = *(const v4f*)(in + i);
        v4h o;
        o[0] = (_Float16)v[0]; o[1] = (_Float16)v[1];
        o[2] = (_Float16)v[2]; o[3] = (_Float16)v[3];
        *(v4h*)(out + i) = o;
    }
}

// ---------------------------------------------------------------------------
// Big GEMM: xprojT[col][row] = bias[col] + sum_k A[row][k] * W[col][k]
// One wave computes a 64x64 block (4x4 WMMA accumulators, K-chunks of 32).
// grid.x = (MTOT/64)*16 = 8192, block = 128.
// ---------------------------------------------------------------------------
__global__ __launch_bounds__(128) void lstm_gemm_xprojT(
        const _Float16* __restrict__ A,       // [MTOT][HID]
        const _Float16* __restrict__ W,       // [GATES][HID]
        const float*    __restrict__ bias,    // [GATES]
        _Float16*       __restrict__ xprojT)  // [GATES][MTOT]
{
    const int lane  = threadIdx.x & 31;
    const int wv    = threadIdx.x >> 5;
    const int m_blk = blockIdx.x >> 4;                   // 0..511
    const int n_blk = ((blockIdx.x & 15) << 2) | wv;     // 0..63
    const int lrow  = lane & 15;
    const int hi    = lane >> 4;

    v8f acc[4][4];
#pragma unroll
    for (int mi = 0; mi < 4; ++mi)
#pragma unroll
        for (int nj = 0; nj < 4; ++nj)
            acc[mi][nj] = (v8f)(0.0f);

    const _Float16* Abase = A + (size_t)(m_blk * 64 + lrow) * HID;
    const _Float16* Wbase = W + (size_t)(n_blk * 64 + lrow) * HID;

    for (int kc = 0; kc < HID; kc += 32) {
        v16h af[4], bf[4];
        const int k0 = kc + hi * 16;
        const int kb = kc + hi * 16;
#pragma unroll
        for (int mi = 0; mi < 4; ++mi)
            af[mi] = load_a_frag(Abase + (size_t)mi * 16 * HID + k0);
#pragma unroll
        for (int nj = 0; nj < 4; ++nj)
            bf[nj] = *(const v16h*)(Wbase + (size_t)nj * 16 * HID + kb);
#pragma unroll
        for (int mi = 0; mi < 4; ++mi)
#pragma unroll
            for (int nj = 0; nj < 4; ++nj)
                acc[mi][nj] = wmma_f16(af[mi], bf[nj], acc[mi][nj]);
    }

#pragma unroll
    for (int nj = 0; nj < 4; ++nj) {
        const int colg = n_blk * 64 + nj * 16 + lrow;
        const float bv = bias[colg];
#pragma unroll
        for (int mi = 0; mi < 4; ++mi) {
            const int rowg = m_blk * 64 + mi * 16 + hi * 8;
            v8h o;
#pragma unroll
            for (int r = 0; r < 8; ++r)
                o[r] = (_Float16)(acc[mi][nj][r] + bv);
            *(v8h*)(xprojT + (size_t)colg * MTOT + rowg) = o;
        }
    }
}

// ---------------------------------------------------------------------------
// Persistent recurrence: one launch per layer, 64 blocks x 128 threads.
// Block = one 16-wide hidden slice (all 4 gate columns staged in LDS once);
// wave = one 16-row batch tile; 4 gate accumulators per wave so the LSTM
// cell update happens entirely in the WMMA C registers. Steps are separated
// by a device-wide release/acquire barrier on a monotonic counter.
// ---------------------------------------------------------------------------
__global__ __launch_bounds__(128) void lstm_recurrence(
        const _Float16* __restrict__ xprojT,   // [GATES][MTOT]
        const _Float16* __restrict__ Whh,      // [GATES][HID]
        _Float16*       __restrict__ h0buf,    // [BATCH][HID] (zeroed)
        _Float16*       __restrict__ h1buf,    // [BATCH][HID]
        float*          __restrict__ c,        // [BATCH][HID] (zeroed)
        float*          __restrict__ out,      // [SEQ][BATCH][HID] f32
        _Float16*       __restrict__ xnext,    // [SEQ*BATCH][HID] f16
        unsigned*       __restrict__ bar)      // zeroed counter
{
    extern __shared__ _Float16 ldsB[];         // 64 cols x LDSPITCH = 132 KB

    const int lane = threadIdx.x & 31;
    const int m    = threadIdx.x >> 5;         // batch tile 0..3
    const int n16  = blockIdx.x;               // hidden tile 0..63
    const int lrow = lane & 15;
    const int hi   = lane >> 4;

    // Stage the block's time-invariant Whh slab (4 gates x 16 cols x 1024 K)
    // into LDS once; the whole 512-step recurrence then reads B via DS.
    for (int idx = threadIdx.x * 8; idx < 64 * HID; idx += 128 * 8) {
        const int cb = idx >> 10;              // 0..63 (gate*16 + col)
        const int k  = idx & 1023;
        const size_t gcol = (size_t)((cb >> 4) * HID + n16 * 16 + (cb & 15));
        v8h v = *(const v8h*)(Whh + gcol * HID + k);
        *(v8h*)(&ldsB[cb * LDSPITCH + k]) = v;
    }
    __syncthreads();

    const unsigned nblocks = gridDim.x;

    for (int t = 0; t < SEQ; ++t) {
        const _Float16* hp = (t & 1) ? h1buf : h0buf;
        _Float16*       hn = (t & 1) ? h0buf : h1buf;

        // Accumulators preloaded with xproj (bias already folded in).
        v8f acc[4];
#pragma unroll
        for (int g = 0; g < 4; ++g) {
            const int colg = (g << 10) + (n16 << 4) + lrow;
            const int rowg = t * BATCH + m * 16 + hi * 8;
            v8h xp = *(const v8h*)(xprojT + (size_t)colg * MTOT + rowg);
#pragma unroll
            for (int r = 0; r < 8; ++r)
                acc[g][r] = (float)xp[r];
        }

        const _Float16* hrow = hp + (size_t)(m * 16 + lrow) * HID;
        for (int kc = 0; kc < HID; kc += 32) {
            v16h a = load_a_frag(hrow + kc + hi * 16);
#pragma unroll
            for (int g = 0; g < 4; ++g) {
                const _Float16* bp =
                    &ldsB[(g * 16 + lrow) * LDSPITCH + kc + hi * 16];
                acc[g] = wmma_f16(a, load_b_frag_lds(bp), acc[g]);
            }
        }

        // LSTM cell update in registers. C elem r -> batch m*16+hi*8+r,
        // hidden n16*16+lrow.
        float* out_t = out + (size_t)t * BATCH * HID;
        _Float16* xn_t = xnext + (size_t)t * BATCH * HID;
#pragma unroll
        for (int r = 0; r < 8; ++r) {
            const int bi  = m * 16 + hi * 8 + r;
            const int ni  = (n16 << 4) + lrow;
            const int idx = bi * HID + ni;
            const float iv = sigmoid_f(acc[0][r]);
            const float fv = sigmoid_f(acc[1][r]);
            const float gv = tanh_f(acc[2][r]);
            const float ov = sigmoid_f(acc[3][r]);
            const float cn = fv * c[idx] + iv * gv;
            c[idx] = cn;
            const float hnv = ov * tanh_f(cn);
            out_t[idx] = hnv;
            const _Float16 hh = (_Float16)hnv;
            hn[idx]   = hh;
            xn_t[idx] = hh;
        }

        // ---- device-wide barrier (release h writes / acquire before reads)
        __threadfence();                       // make this thread's stores visible
        __syncthreads();
        if (threadIdx.x == 0)
            __hip_atomic_fetch_add(bar, 1u, __ATOMIC_RELEASE,
                                   __HIP_MEMORY_SCOPE_AGENT);
        const unsigned target = nblocks * (unsigned)(t + 1);
        while (__hip_atomic_load(bar, __ATOMIC_ACQUIRE,
                                 __HIP_MEMORY_SCOPE_AGENT) < target)
            __builtin_amdgcn_s_sleep(2);
        __syncthreads();
    }
}

// ---------------------------------------------------------------------------
// Host orchestration
// ---------------------------------------------------------------------------
extern "C" void kernel_launch(void* const* d_in, const int* in_sizes, int n_in,
                              void* d_out, int out_size, void* d_ws, size_t ws_size,
                              hipStream_t stream) {
    (void)in_sizes; (void)n_in; (void)out_size;

    const float* x    = (const float*)d_in[0];   // [512,64,1024]
    const float* Wih  = (const float*)d_in[1];   // [2,4096,1024]
    const float* Whh  = (const float*)d_in[2];   // [2,4096,1024]
    const float* bias = (const float*)d_in[3];   // [2,4096]

    float* out = (float*)d_out;                          // [512,64,1024]
    float* h_n = out + (size_t)SEQ * BATCH * HID;        // [2,64,1024]
    float* c_n = h_n + (size_t)2 * BATCH * HID;          // [2,64,1024]

    char* ws = (char*)d_ws;
    size_t off = 0;
    _Float16* x16    = (_Float16*)(ws + off); off += (size_t)MTOT  * HID * 2;  // 64 MB
    _Float16* wih16  = (_Float16*)(ws + off); off += (size_t)GATES * HID * 2;  // 8 MB
    _Float16* whh16  = (_Float16*)(ws + off); off += (size_t)GATES * HID * 2;  // 8 MB
    _Float16* xprojT = (_Float16*)(ws + off); off += (size_t)GATES * MTOT * 2; // 256 MB
    _Float16* h0     = (_Float16*)(ws + off); off += (size_t)BATCH * HID * 2;
    _Float16* h1     = (_Float16*)(ws + off); off += (size_t)BATCH * HID * 2;
    float*    cbuf   = (float*)   (ws + off); off += (size_t)BATCH * HID * 4;
    unsigned* bar    = (unsigned*)(ws + off); off += 256;
    if (ws_size < off) return;  // workspace too small; nothing safe to do

    const int ldsBytes = 64 * LDSPITCH * 2;    // 132096 B (<320KB per WGP)
    (void)hipFuncSetAttribute((const void*)lstm_recurrence,
                              hipFuncAttributeMaxDynamicSharedMemorySize,
                              ldsBytes);

    // Layer-0 input -> f16
    {
        const int n = MTOT * HID;
        cvt_f32_to_f16<<<(n / 4 + 255) / 256, 256, 0, stream>>>(x, x16, n);
    }

    for (int layer = 0; layer < 2; ++layer) {
        const int nw = GATES * HID;
        cvt_f32_to_f16<<<(nw / 4 + 255) / 256, 256, 0, stream>>>(
            Wih + (size_t)layer * nw, wih16, nw);
        cvt_f32_to_f16<<<(nw / 4 + 255) / 256, 256, 0, stream>>>(
            Whh + (size_t)layer * nw, whh16, nw);

        // xprojT = (layer input) @ Wih^T + b, stored transposed [4H][T*B]
        lstm_gemm_xprojT<<<(MTOT / 64) * 16, 128, 0, stream>>>(
            x16, wih16, bias + (size_t)layer * GATES, xprojT);

        hipMemsetAsync(h0,   0, (size_t)BATCH * HID * 2, stream);
        hipMemsetAsync(cbuf, 0, (size_t)BATCH * HID * 4, stream);
        hipMemsetAsync(bar,  0, sizeof(unsigned), stream);

        // Whole 512-step recurrence in one persistent launch (64 blocks,
        // all co-resident; device-wide barrier per step).
        lstm_recurrence<<<HID / 16, 128, ldsBytes, stream>>>(
            xprojT, whh16, h0, h1, cbuf, out, x16, bar);

        hipMemcpyAsync(h_n + (size_t)layer * BATCH * HID,
                       out + (size_t)(SEQ - 1) * BATCH * HID,
                       (size_t)BATCH * HID * 4, hipMemcpyDeviceToDevice, stream);
        hipMemcpyAsync(c_n + (size_t)layer * BATCH * HID, cbuf,
                       (size_t)BATCH * HID * 4, hipMemcpyDeviceToDevice, stream);
    }
}
